// ImprovedTransformerGNN_43121471652515
// MI455X (gfx1250) — matrix operand order repaired and checked by reading, not compiled, not run on hardware
//
#include <hip/hip_runtime.h>
#include <math.h>

// ---------------- problem constants ----------------
#define HDIM 128
#define GNUM 128
#define CNUM 4

typedef _Float16 half16 __attribute__((ext_vector_type(16)));
typedef float float8 __attribute__((ext_vector_type(8)));
typedef unsigned int u32x4 __attribute__((ext_vector_type(4)));
typedef int i32x4 __attribute__((ext_vector_type(4)));
typedef int i32x8 __attribute__((ext_vector_type(8)));

__device__ __forceinline__ float gelu_f(float x) {
    // exact GELU: x * 0.5 * (1 + erf(x / sqrt(2)))
    return 0.5f * x * (1.0f + erff(x * 0.70710678118654752440f));
}
#define BN_SC 0.99999500003749978f  // 1/sqrt(1+1e-5)

__device__ __forceinline__ void atomicMaxF(float* addr, float val) {
    if (val >= 0.0f) atomicMax((int*)addr, __float_as_int(val));
    else             atomicMin((unsigned int*)addr, __float_as_uint(val));
}

#define TDM_AVAIL (__has_builtin(__builtin_amdgcn_tensor_load_to_lds) && \
                   __has_builtin(__builtin_amdgcn_s_wait_tensorcnt))

// ---------------------------------------------------------------------------
// Weight pre-pass: TDM the f32 weight matrix [KD x HOUT] into LDS, then emit a
// transposed f16 copy WTg[n*KD + k] to global workspace. One block per matrix.
// ---------------------------------------------------------------------------
template<int KD, int HOUT>
__global__ __launch_bounds__(256)
void k_wconv(const float* __restrict__ W, _Float16* __restrict__ WTg)
{
#if TDM_AVAIL
    __shared__ float S[KD * HOUT];  // 128x128 f32 = 64KB max
    if ((threadIdx.x >> 5) == 0) {  // one wave issues the DMA (EXEC ignored)
        uint64_t ga = (uint64_t)(uintptr_t)W;
        uint32_t lds = (uint32_t)(uintptr_t)(&S[0]);  // flat LDS addr low 32 = offset
        // D# group 0: count=1 | lds_addr | global_addr[56:0] | type=2
        u32x4 g0;
        g0[0] = 1u;
        g0[1] = lds;
        g0[2] = (uint32_t)ga;
        g0[3] = (uint32_t)((ga >> 32) & 0x01FFFFFFu) | (2u << 30);
        // D# group 1: data_size=4B, dims: dim0=HOUT (contig), dim1=KD,
        // tile = full matrix, stride0 = HOUT
        i32x8 g1;
        g1[0] = (int)(2u << 16);                                          // data_size=2 (4B)
        g1[1] = (int)(((unsigned)HOUT & 0xFFFFu) << 16);                  // dim0[15:0] @ bit48
        g1[2] = (int)((((unsigned)HOUT >> 16) & 0xFFFFu) |
                      (((unsigned)KD & 0xFFFFu) << 16));                  // dim0 hi | dim1 lo
        g1[3] = (int)((((unsigned)KD >> 16) & 0xFFFFu) |
                      (((unsigned)HOUT & 0xFFFFu) << 16));                // dim1 hi | tile_dim0
        g1[4] = (int)((unsigned)KD & 0xFFFFu);                            // tile_dim1; tile_dim2=0
        g1[5] = (int)(unsigned)HOUT;                                      // stride0[31:0]
        g1[6] = 0;
        g1[7] = 0;
        i32x4 z4 = {0, 0, 0, 0};
#if __clang_major__ <= 22
        __builtin_amdgcn_tensor_load_to_lds(g0, g1, z4, z4, 0);
#else
        i32x8 z8 = {0, 0, 0, 0, 0, 0, 0, 0};
        __builtin_amdgcn_tensor_load_to_lds(g0, g1, z4, z4, z8, 0);
#endif
        __builtin_amdgcn_s_wait_tensorcnt(0);
    }
    __syncthreads();
    for (int idx = threadIdx.x; idx < KD * HOUT; idx += 256) {
        int n = idx % HOUT, k = idx / HOUT;   // coalesced LDS read side
        WTg[(size_t)n * KD + k] = (_Float16)S[idx];
    }
#else
    for (int idx = threadIdx.x; idx < KD * HOUT; idx += 256) {
        int n = idx % HOUT, k = idx / HOUT;
        WTg[(size_t)n * KD + k] = (_Float16)W[idx];
    }
#endif
}

// ---------------------------------------------------------------------------
// WMMA GEMM, pre-transposed f16 weights staged through LDS.
// out = epilogue( in[nrows x KD] @ W[KD x NT*16] )
// EP: 1 = +bias,gelu -> f32 | 2 = +bias,BN,gelu -> f32
//     3 = GCN pre-agg: out32 = acc*dinv[row]^2 (self loop), out16 = (f16)acc
// block = 256 threads (8 waves); wave = 16 rows; block = 128 rows.
// ---------------------------------------------------------------------------
template<int KD, int NT, int EP>
__global__ __launch_bounds__(256)
void gemm_lds(const float* __restrict__ in, const _Float16* __restrict__ WTg,
              const float* __restrict__ bias, const float* __restrict__ gamma,
              const float* __restrict__ beta, float* __restrict__ out32,
              _Float16* __restrict__ out16, const float* __restrict__ dinv, int nrows)
{
    constexpr int HOUT = NT * 16;
    constexpr int KP = KD + 8;         // 272B row stride -> 16B aligned
    __shared__ _Float16 WT[HOUT][KP];  // WT[n][k]

    for (int idx = threadIdx.x; idx < HOUT * KD; idx += 256) {
        int n = idx / KD, k = idx % KD;
        WT[n][k] = WTg[idx];
    }
    __syncthreads();

    const int lane = threadIdx.x & 31;
    const int wave = threadIdx.x >> 5;
    const int rowBase = blockIdx.x * 128 + wave * 16;

    int m = rowBase + (lane & 15);
    if (m > nrows - 1) m = nrows - 1;
    const float* __restrict__ inrow = in + (size_t)m * KD;

    float8 acc[NT] = {};
    const int kaLo = (lane >> 4) << 4;   // A: lanes 16-31 hold K+16
    const int kbB  = (lane >> 4) << 4;   // B: lanes 16-31 hold K+16
    const int ncol = lane & 15;

#pragma unroll
    for (int kc = 0; kc < KD / 32; ++kc) {
        half16 a;
        const int kb = kc * 32 + kaLo;
#pragma unroll
        for (int j = 0; j < 8; ++j) a[j] = (_Float16)inrow[kb + j];
#pragma unroll
        for (int j = 0; j < 8; ++j) a[8 + j] = (_Float16)inrow[kb + 8 + j];

#pragma unroll
        for (int nt = 0; nt < NT; ++nt) {
            half16 b;
            const _Float16* __restrict__ bp = &WT[nt * 16 + ncol][kc * 32 + kbB];
#pragma unroll
            for (int j = 0; j < 16; ++j) b[j] = bp[j];
            acc[nt] = __builtin_amdgcn_wmma_f32_16x16x32_f16(
                false, a, false, b, (short)0, acc[nt], false, false);
        }
    }

    // C/D layout: VGPR r -> row r + 8*(lane>>4), col = lane&15
    const int rOff = (lane >> 4) << 3;
    if constexpr (EP == 3) {
#pragma unroll
        for (int r = 0; r < 8; ++r) {
            const int row = rowBase + rOff + r;
            if (row < nrows) {
                const float di = dinv[row];
                const float sn = di * di;
#pragma unroll
                for (int nt = 0; nt < NT; ++nt) {
                    const int col = nt * 16 + ncol;
                    const float v = acc[nt][r];
                    out32[(size_t)row * HOUT + col] = v * sn;   // agg self-loop init
                    out16[(size_t)row * HOUT + col] = (_Float16)v;  // xw for edge gather
                }
            }
        }
    } else {
#pragma unroll
        for (int nt = 0; nt < NT; ++nt) {
            const int col = nt * 16 + ncol;
            float bia = 0.f, gsc = 0.f, bet = 0.f;
            if constexpr (EP >= 1) bia = bias[col];
            if constexpr (EP == 2) { gsc = gamma[col] * BN_SC; bet = beta[col]; }
#pragma unroll
            for (int r = 0; r < 8; ++r) {
                const int row = rowBase + rOff + r;
                if (row < nrows) {
                    float v = acc[nt][r];
                    if constexpr (EP >= 1) v += bia;
                    if constexpr (EP == 2) v = v * gsc + bet;
                    if constexpr (EP >= 1) v = gelu_f(v);
                    out32[(size_t)row * HOUT + col] = v;
                }
            }
        }
    }
}

// ---------------------------------------------------------------------------
// WMMA GEMM, f32 B-fragments straight from global (tiny classifier matrices).
// ---------------------------------------------------------------------------
template<int KD, int NT, int EP>
__global__ __launch_bounds__(256)
void gemm_glob(const float* __restrict__ in, const float* __restrict__ W,
               const float* __restrict__ bias, const float* __restrict__ gamma,
               const float* __restrict__ beta, float* __restrict__ out, int nrows)
{
    constexpr int HOUT = NT * 16;
    const int lane = threadIdx.x & 31;
    const int wave = threadIdx.x >> 5;
    const int rowBase = blockIdx.x * 128 + wave * 16;

    int m = rowBase + (lane & 15);
    if (m > nrows - 1) m = nrows - 1;
    const float* __restrict__ inrow = in + (size_t)m * KD;

    float8 acc[NT] = {};
    const int kaLo = (lane >> 4) << 4;
    const int kbB  = (lane >> 4) << 4;
    const int ncol = lane & 15;

    for (int kc = 0; kc < KD / 32; ++kc) {
        half16 a;
        const int kb = kc * 32 + kaLo;
#pragma unroll
        for (int j = 0; j < 8; ++j) a[j] = (_Float16)inrow[kb + j];
#pragma unroll
        for (int j = 0; j < 8; ++j) a[8 + j] = (_Float16)inrow[kb + 8 + j];

#pragma unroll
        for (int nt = 0; nt < NT; ++nt) {
            half16 b;
            const int n = nt * 16 + ncol;
            const int k0 = kc * 32 + kbB;
#pragma unroll
            for (int j = 0; j < 16; ++j)
                b[j] = (_Float16)W[(size_t)(k0 + j) * HOUT + n];
            acc[nt] = __builtin_amdgcn_wmma_f32_16x16x32_f16(
                false, a, false, b, (short)0, acc[nt], false, false);
        }
    }

    const int rOff = (lane >> 4) << 3;
#pragma unroll
    for (int nt = 0; nt < NT; ++nt) {
        const int col = nt * 16 + ncol;
        float bia = 0.f, gsc = 0.f, bet = 0.f;
        if constexpr (EP >= 1) bia = bias[col];
        if constexpr (EP == 2) { gsc = gamma[col] * BN_SC; bet = beta[col]; }
#pragma unroll
        for (int r = 0; r < 8; ++r) {
            const int row = rowBase + rOff + r;
            if (row < nrows) {
                float v = acc[nt][r];
                if constexpr (EP >= 1) v += bia;
                if constexpr (EP == 2) v = v * gsc + bet;
                if constexpr (EP >= 1) v = gelu_f(v);
                out[(size_t)row * HOUT + col] = v;
            }
        }
    }
}

// ---------------- small helper kernels ----------------
__global__ void k_zero(float* p, int n) {
    int i = blockIdx.x * blockDim.x + threadIdx.x;
    if (i < n) p[i] = 0.0f;
}

__global__ void k_deg(const int* __restrict__ dst, float* __restrict__ deg, int E) {
    int i = blockIdx.x * blockDim.x + threadIdx.x;
    if (i < E) atomicAdd(&deg[dst[i]], 1.0f);
}

__global__ void k_dinv(float* d, int n) {
    int i = blockIdx.x * blockDim.x + threadIdx.x;
    if (i < n) d[i] = rsqrtf(d[i] + 1.0f);
}

// one wave per edge; each lane handles 4 coalesced features (f16 gather)
__global__ void k_edgeagg(const int* __restrict__ src, const int* __restrict__ dst,
                          const float* __restrict__ dinv, const _Float16* __restrict__ xw,
                          float* __restrict__ agg, int E) {
    int t = blockIdx.x * blockDim.x + threadIdx.x;
    int e = t >> 5;
    int lane = t & 31;
    if (e < E) {
        int s = src[e], d = dst[e];
        float w = dinv[s] * dinv[d];
        const _Float16* __restrict__ xs = xw + (size_t)s * HDIM;
        float* __restrict__ ad = agg + (size_t)d * HDIM;
        __builtin_prefetch(xs, 0, 1);
#pragma unroll
        for (int c = 0; c < 4; ++c) {
            int f = lane + 32 * c;
            atomicAdd(ad + f, (float)xs[f] * w);
        }
    }
}

// h = gelu(bn(agg + bg)) + h   (in-place residual, elementwise-safe)
__global__ void k_post(const float* __restrict__ agg, const float* __restrict__ bg,
                       const float* __restrict__ gg, const float* __restrict__ bgg,
                       float* __restrict__ h, int total) {
    int i = blockIdx.x * blockDim.x + threadIdx.x;
    if (i < total) {
        int f = i & (HDIM - 1);
        float v = (agg[i] + bg[f]) * (gg[f] * BN_SC) + bgg[f];
        h[i] = gelu_f(v) + h[i];
    }
}

// aw = tanh(t @ Wa2 + ba2), t: [N,64] f32
__global__ void k_att2(const float* __restrict__ t, const float* __restrict__ Wa2,
                       const float* __restrict__ ba2, float* __restrict__ aw, int N) {
    int i = blockIdx.x * blockDim.x + threadIdx.x;
    if (i < N) {
        float s = ba2[0];
        const float* tr = t + (size_t)i * 64;
#pragma unroll 8
        for (int k = 0; k < 64; ++k) s += tr[k] * Wa2[k];
        aw[i] = tanhf(s);
    }
}

__global__ void k_poolinit(float* __restrict__ psum, float* __restrict__ pmax,
                           float* __restrict__ pw, float* __restrict__ cnt, int gh) {
    int i = blockIdx.x * blockDim.x + threadIdx.x;
    if (i < gh) {
        psum[i] = 0.0f;
        pw[i] = 0.0f;
        pmax[i] = -3.402823466e38f;
        if (i < GNUM) cnt[i] = 0.0f;
    }
}

__global__ void k_cnt(const int* __restrict__ batch, float* __restrict__ cnt, int N) {
    int i = blockIdx.x * blockDim.x + threadIdx.x;
    if (i < N) atomicAdd(&cnt[batch[i]], 1.0f);
}

__global__ void k_pool(const float* __restrict__ h, const float* __restrict__ aw,
                       const int* __restrict__ batch, float* __restrict__ psum,
                       float* __restrict__ pmax, float* __restrict__ pw, int total) {
    int i = blockIdx.x * blockDim.x + threadIdx.x;
    if (i < total) {
        int node = i >> 7, f = i & (HDIM - 1);
        int g = batch[node];
        float v = h[i];
        size_t o = (size_t)g * HDIM + f;
        atomicAdd(&psum[o], v);
        atomicMaxF(&pmax[o], v);
        atomicAdd(&pw[o], v * aw[node]);
    }
}

// comb[g, 0:512] = [mean | max | mean | weighted]
__global__ void k_comb(const float* __restrict__ psum, const float* __restrict__ pmax,
                       const float* __restrict__ pw, const float* __restrict__ cnt,
                       float* __restrict__ comb, int total) {
    int i = blockIdx.x * blockDim.x + threadIdx.x;
    if (i < total) {
        int g = i >> 7, f = i & (HDIM - 1);
        float c = fmaxf(cnt[g], 1.0f);
        float mean = psum[i] / c;
        float* cb = comb + (size_t)g * (4 * HDIM);
        cb[f] = mean;
        cb[HDIM + f] = pmax[i];
        cb[2 * HDIM + f] = mean;
        cb[3 * HDIM + f] = pw[i] / c;
    }
}

// final: out[g,c] = z3[g,:] @ Wc4 + bc4
__global__ void k_final(const float* __restrict__ z, const float* __restrict__ W,
                        const float* __restrict__ b, float* __restrict__ out, int total) {
    int i = blockIdx.x * blockDim.x + threadIdx.x;
    if (i < total) {
        int g = i >> 2, c = i & 3;
        float s = b[c];
        const float* zr = z + (size_t)g * 64;
#pragma unroll 8
        for (int k = 0; k < 64; ++k) s += zr[k] * W[k * CNUM + c];
        out[i] = s;
    }
}

// ---------------------------------------------------------------------------
extern "C" void kernel_launch(void* const* d_in, const int* in_sizes, int n_in,
                              void* d_out, int out_size, void* d_ws, size_t ws_size,
                              hipStream_t stream) {
    const float* x    = (const float*)d_in[0];
    const int*   edge = (const int*)d_in[1];
    const int*   batch= (const int*)d_in[2];
    const float* W1 = (const float*)d_in[3];  const float* b1  = (const float*)d_in[4];
    const float* g1 = (const float*)d_in[5];  const float* be1 = (const float*)d_in[6];
    const float* W2 = (const float*)d_in[7];  const float* b2  = (const float*)d_in[8];
    const float* g2 = (const float*)d_in[9];  const float* be2 = (const float*)d_in[10];
    const float* Wg = (const float*)d_in[11]; const float* bg  = (const float*)d_in[12];
    const float* gg = (const float*)d_in[13]; const float* bgg = (const float*)d_in[14];
    const float* Wa1= (const float*)d_in[15]; const float* ba1 = (const float*)d_in[16];
    const float* Wa2= (const float*)d_in[17]; const float* ba2 = (const float*)d_in[18];
    const float* Wc1= (const float*)d_in[19]; const float* bc1 = (const float*)d_in[20];
    const float* gc1= (const float*)d_in[21]; const float* bec1= (const float*)d_in[22];
    const float* Wc2= (const float*)d_in[23]; const float* bc2 = (const float*)d_in[24];
    const float* gc2= (const float*)d_in[25]; const float* bec2= (const float*)d_in[26];
    const float* Wc3= (const float*)d_in[27]; const float* bc3 = (const float*)d_in[28];
    const float* gc3= (const float*)d_in[29]; const float* bec3= (const float*)d_in[30];
    const float* Wc4= (const float*)d_in[31]; const float* bc4 = (const float*)d_in[32];
    float* out = (float*)d_out;

    const int N = in_sizes[0] / HDIM;
    const int E = in_sizes[1] / 2;
    const int* src = edge;
    const int* dst = edge + E;
    const size_t NH = (size_t)N * HDIM;

    // ---- workspace carve-up ----
    char* base = (char*)d_ws;
    // pre-transposed f16 weights: W1,W2,Wg0..2 (128x128 each), Wa1 (128x64)
    _Float16* W1T  = (_Float16*)base;            base += 16384 * 2;
    _Float16* W2T  = (_Float16*)base;            base += 16384 * 2;
    _Float16* WgT0 = (_Float16*)base;            base += 16384 * 2;
    _Float16* WgT1 = (_Float16*)base;            base += 16384 * 2;
    _Float16* WgT2 = (_Float16*)base;            base += 16384 * 2;
    _Float16* Wa1T = (_Float16*)base;            base += 8192 * 2;
    base = (char*)(((uintptr_t)base + 255) & ~(uintptr_t)255);
    float*     buf_h = (float*)base;             base += NH * 4;   // node features
    float*     agg   = (float*)base;             base += NH * 4;   // agg / proj scratch
    _Float16*  xw16  = (_Float16*)base;                            // xw (f16) ...
    float*     att   = (float*)base;             base += NH * 2;   // ... aliases att [N,64] f32
    float*     dinv  = (float*)base;             base += (size_t)N * 4;
    float*     aw    = (float*)base;             base += (size_t)N * 4;
    float*     cnt   = (float*)base;             base += GNUM * 4;
    float*     psum  = (float*)base;             base += GNUM * HDIM * 4;
    float*     pmax  = (float*)base;             base += GNUM * HDIM * 4;
    float*     pwsum = (float*)base;             base += GNUM * HDIM * 4;
    float*     comb  = (float*)base;             base += GNUM * 4 * HDIM * 4;
    float*     z1    = (float*)base;             base += GNUM * 256 * 4;
    float*     z2    = (float*)base;             base += GNUM * HDIM * 4;
    float*     z3    = (float*)base;             base += GNUM * 64 * 4;

    const dim3 blk(256);
    const int gGemm = (N + 127) / 128;
    const int gN    = (N + 255) / 256;
    const int gE    = (E + 255) / 256;
    const int gNH   = (int)((NH + 255) / 256);
    const int gEw   = (int)(((size_t)E * 32 + 255) / 256);

    // ---- weight pre-convert (TDM -> LDS -> transposed f16 in workspace) ----
    k_wconv<128, 128><<<1, blk, 0, stream>>>(W1, W1T);
    k_wconv<128, 128><<<1, blk, 0, stream>>>(W2, W2T);
    k_wconv<128, 128><<<1, blk, 0, stream>>>(Wg, WgT0);
    k_wconv<128, 128><<<1, blk, 0, stream>>>(Wg + 16384, WgT1);
    k_wconv<128, 128><<<1, blk, 0, stream>>>(Wg + 32768, WgT2);
    k_wconv<128, 64><<<1, blk, 0, stream>>>(Wa1, Wa1T);

    // ---- degrees / symmetric norm ----
    k_zero<<<gN, blk, 0, stream>>>(dinv, N);
    k_deg<<<gE, blk, 0, stream>>>(dst, dinv, E);
    k_dinv<<<gN, blk, 0, stream>>>(dinv, N);

    // ---- input projection: h = gelu(bn(gelu(bn(x@W1))@W2)) ----
    gemm_lds<128, 8, 2><<<gGemm, blk, 0, stream>>>(x, W1T, b1, g1, be1, agg, nullptr, nullptr, N);
    gemm_lds<128, 8, 2><<<gGemm, blk, 0, stream>>>(agg, W2T, b2, g2, be2, buf_h, nullptr, nullptr, N);

    // ---- L=3 GCN layers ----
    _Float16* WgT[3] = {WgT0, WgT1, WgT2};
    for (int l = 0; l < 3; ++l) {
        const float* bl  = bg  + (size_t)l * HDIM;
        const float* gl  = gg  + (size_t)l * HDIM;
        const float* bbl = bgg + (size_t)l * HDIM;
        // xw + fused self-loop agg init (agg = xw*dinv^2, xw16 = f16 copy)
        gemm_lds<128, 8, 3><<<gGemm, blk, 0, stream>>>(buf_h, WgT[l], nullptr, nullptr, nullptr,
                                                       agg, xw16, dinv, N);
        k_edgeagg<<<gEw, blk, 0, stream>>>(src, dst, dinv, xw16, agg, E);
        k_post<<<gNH, blk, 0, stream>>>(agg, bl, gl, bbl, buf_h, (int)NH);
    }

    // ---- attention scores ----
    gemm_lds<128, 4, 1><<<gGemm, blk, 0, stream>>>(buf_h, Wa1T, ba1, nullptr, nullptr,
                                                   att, nullptr, nullptr, N);
    k_att2<<<gN, blk, 0, stream>>>(att, Wa2, ba2, aw, N);

    // ---- per-graph pooling ----
    k_poolinit<<<(GNUM * HDIM + 255) / 256, blk, 0, stream>>>(psum, pmax, pwsum, cnt, GNUM * HDIM);
    k_cnt<<<gN, blk, 0, stream>>>(batch, cnt, N);
    k_pool<<<gNH, blk, 0, stream>>>(buf_h, aw, batch, psum, pmax, pwsum, (int)NH);
    k_comb<<<(GNUM * 4 * HDIM + 255) / 256, blk, 0, stream>>>(psum, pmax, pwsum, cnt, comb, GNUM * HDIM);

    // ---- classifier ----
    gemm_glob<512, 16, 2><<<1, blk, 0, stream>>>(comb, Wc1, bc1, gc1, bec1, z1, GNUM);
    gemm_glob<256, 8, 2><<<1, blk, 0, stream>>>(z1, Wc2, bc2, gc2, bec2, z2, GNUM);
    gemm_glob<128, 4, 2><<<1, blk, 0, stream>>>(z2, Wc3, bc3, gc3, bec3, z3, GNUM);
    k_final<<<(GNUM * CNUM + 255) / 256, blk, 0, stream>>>(z3, Wc4, bc4, out, GNUM * CNUM);
}
